// UpSampler_27315992003060
// MI455X (gfx1250) — matrix operand order repaired and checked by reading, not compiled, not run on hardware
//
#include <hip/hip_runtime.h>

typedef __attribute__((ext_vector_type(16))) _Float16 v16h;
typedef __attribute__((ext_vector_type(8)))  _Float16 v8h;
typedef __attribute__((ext_vector_type(8)))  float    v8f;

#define WAVES_PER_BLOCK 8
#define CONV_TM 4   // 16-voxel tiles per wave in the conv kernels

// ---------------------------------------------------------------------------
// Fragment helpers (layouts per CDNA5 ISA 7.12.2, wave32):
//  A 16x32 f16 : lane row = lane&15; lanes<16 hold K {0..7,16..23}, lanes>=16
//                hold K {8..15,24..31}  -> two contiguous 8-half runs per lane.
//  B 32x16 f16 : lane col = lane&15; lanes<16 hold K {0..7,16..23}, lanes>=16 K {8..15,24..31}
//                -> prepacked so each lane reads 16 contiguous halves.
//  C/D 16x16 f32: lane col = lane&15; vgpr j -> row j + 8*(lane>>4).
// ---------------------------------------------------------------------------

static __device__ __forceinline__ v16h load_afrag(const _Float16* p, int hi) {
  const v8h* q = (const v8h*)(p + hi * 8);
  v8h lo = q[0];   // K = base0 .. base0+7
  v8h hh = q[2];   // K = base0+16 .. base0+23
  v16h a;
#pragma unroll
  for (int i = 0; i < 8; ++i) { a[i] = lo[i]; a[i + 8] = hh[i]; }
  return a;
}

static __device__ __forceinline__ v16h load_bfrag(const _Float16* Wf, int frag, int lane) {
  return *(const v16h*)(Wf + ((size_t)frag << 9) + ((size_t)lane << 4));
}

static __device__ __forceinline__ v8f wmma_f16(v16h a, v16h b, v8f c) {
  return __builtin_amdgcn_wmma_f32_16x16x32_f16(false, a, false, b, (short)0, c,
                                                false, false);
}

// ---------------------------------------------------------------------------
// Repack f32 weights [KO][CIN][COUT] into per-lane B-fragment order (f16).
// frag id = (ko*KCN + kc)*NTN + nt ; within frag: lane*16 + e,
// value = W[ko][kc*32 + 16*(e>>3) + 8*(lane>>4) + (e&7)][nt*16 + (lane&15)]
// ---------------------------------------------------------------------------
__global__ void repack_weights_kernel(const float* __restrict__ W,
                                      _Float16* __restrict__ out,
                                      int KO, int CIN, int COUT) {
  const int ntn = COUT >> 4, kcn = CIN >> 5;
  const int total = KO * kcn * ntn * 512;
  for (int tid = blockIdx.x * blockDim.x + threadIdx.x; tid < total;
       tid += gridDim.x * blockDim.x) {
    int f = tid >> 9, r = tid & 511;
    int lane = r >> 4, e = r & 15;
    int nt = f % ntn;
    int t1 = f / ntn;
    int kc = t1 % kcn;
    int ko = t1 / kcn;
    int K = kc * 32 + ((e >> 3) << 4) + ((lane >> 4) << 3) + (e & 7);
    int n = nt * 16 + (lane & 15);
    out[tid] = (_Float16)W[((size_t)ko * CIN + K) * COUT + n];
  }
}

__global__ void cvt_feats_kernel(const float* __restrict__ X,
                                 _Float16* __restrict__ Y, int N, int C) {
  size_t total = (size_t)(N + 1) * C;
  for (size_t tid = (size_t)blockIdx.x * blockDim.x + threadIdx.x; tid < total;
       tid += (size_t)gridDim.x * blockDim.x) {
    size_t rowi = tid / (size_t)C;
    Y[tid] = (rowi < (size_t)N) ? (_Float16)X[tid] : (_Float16)0.f;
  }
}

__global__ void zero_sentinel_kernel(_Float16* a, _Float16* b, _Float16* c) {
  int t = threadIdx.x;
  if (t < 64) {
    a[t] = (_Float16)0.f;
    b[t] = (_Float16)0.f;
    c[t] = (_Float16)0.f;
  }
}

// ---------------------------------------------------------------------------
// Sparse 3x3x3 conv: one wave -> CONV_TM tiles of 16 voxels x 64 outputs.
// Each B fragment load is amortized over CONV_TM A tiles (weight-traffic /TM).
// out = relu(b + sum_{ko<27} X[nbr[v,ko]] @ W[ko]) ; sentinel row N is zero.
// ---------------------------------------------------------------------------
template <int CIN, int TM>
__global__ __launch_bounds__(256) void conv3x3_kernel(
    const _Float16* __restrict__ X, const int* __restrict__ nbr,
    const _Float16* __restrict__ Wf, const float* __restrict__ bias,
    _Float16* __restrict__ H, int N) {
  const int lane = threadIdx.x & 31;
  const int wid = threadIdx.x >> 5;
  const int tile0 = (blockIdx.x * WAVES_PER_BLOCK + wid) * TM;
  if (tile0 * 16 >= N) return;
  const int row = lane & 15;
  const int hi = lane >> 4;
  constexpr int KCN = CIN / 32;

  v8f c[TM][4];
#pragma unroll
  for (int nt = 0; nt < 4; ++nt) {
    float bv = bias[nt * 16 + row];
#pragma unroll
    for (int t = 0; t < TM; ++t)
#pragma unroll
      for (int j = 0; j < 8; ++j) c[t][nt][j] = bv;
  }

  int base_[TM];
  bool valid[TM];
  const int* nrow[TM];
#pragma unroll
  for (int t = 0; t < TM; ++t) {
    base_[t] = (tile0 + t) << 4;
    int myvox = base_[t] + row;
    valid[t] = myvox < N;
    int vcl = valid[t] ? myvox : (N - 1);
    nrow[t] = nbr + (size_t)vcl * 27;
  }

  for (int ko = 0; ko < 27; ++ko) {
    int idx[TM];
#pragma unroll
    for (int t = 0; t < TM; ++t) idx[t] = valid[t] ? nrow[t][ko] : N;
#pragma unroll
    for (int kc = 0; kc < KCN; ++kc) {
      v16h a[TM];
#pragma unroll
      for (int t = 0; t < TM; ++t)
        a[t] = load_afrag(X + (size_t)idx[t] * CIN + kc * 32, hi);
#pragma unroll
      for (int nt = 0; nt < 4; ++nt) {
        v16h b = load_bfrag(Wf, (ko * KCN + kc) * 4 + nt, lane);
#pragma unroll
        for (int t = 0; t < TM; ++t) c[t][nt] = wmma_f16(a[t], b, c[t][nt]);
      }
    }
  }

#pragma unroll
  for (int t = 0; t < TM; ++t)
#pragma unroll
    for (int nt = 0; nt < 4; ++nt)
#pragma unroll
      for (int j = 0; j < 8; ++j) {
        int r = base_[t] + j + 8 * hi;
        if (r < N) {
          float f = c[t][nt][j];
          f = f > 0.f ? f : 0.f;
          H[(size_t)r * 64 + nt * 16 + row] = (_Float16)f;
        }
      }
}

// ---------------------------------------------------------------------------
// Fused: transposed 2x2x2 conv (per-row weight via per-lane A masking) +
// decoder (relu -> 64x64 -> relu -> 64x32) + residual add. One wave per
// 16-fine-voxel tile; LDS used to re-layout C-fragments into A-fragments.
// ---------------------------------------------------------------------------
__global__ __launch_bounds__(256) void upsample_decode_kernel(
    const _Float16* __restrict__ H3, const int* __restrict__ parent,
    const int* __restrict__ offs, const _Float16* __restrict__ Wtf,
    const float* __restrict__ bt, const _Float16* __restrict__ Wd1f,
    const float* __restrict__ bd1, const _Float16* __restrict__ Wd2f,
    const float* __restrict__ bd2, const float* __restrict__ xup,
    float* __restrict__ out, int M, int N) {
  __shared__ _Float16 lds[WAVES_PER_BLOCK][16 * 64];
  const int lane = threadIdx.x & 31;
  const int wid = threadIdx.x >> 5;
  const int tile = blockIdx.x * WAVES_PER_BLOCK + wid;
  const int base = tile << 4;
  if (base >= M) return;
  const int row = lane & 15;
  const int hi = lane >> 4;

  const int m = base + row;
  const int msafe = m < M ? m : (M - 1);
  const int p = (m < M) ? parent[msafe] : N;   // sentinel -> zero row
  const int off = offs[msafe];

  const _Float16* g = H3 + (size_t)p * 64;
  v16h a0 = load_afrag(g, hi);
  v16h a1 = load_afrag(g + 32, hi);

  v16h z16;
#pragma unroll
  for (int i = 0; i < 16; ++i) z16[i] = (_Float16)0.f;

  // ---- transposed conv: t = b_t + g @ W_t[off] (per-lane row masking) ----
  v8f c[4];
#pragma unroll
  for (int nt = 0; nt < 4; ++nt) {
    float bv = bt[nt * 16 + row];
#pragma unroll
    for (int j = 0; j < 8; ++j) c[nt][j] = bv;
  }
  for (int k = 0; k < 8; ++k) {
    bool sel = (off == k);
    v16h m0 = sel ? a0 : z16;
    v16h m1 = sel ? a1 : z16;
#pragma unroll
    for (int nt = 0; nt < 4; ++nt) {
      c[nt] = wmma_f16(m0, load_bfrag(Wtf, (k * 2 + 0) * 4 + nt, lane), c[nt]);
      c[nt] = wmma_f16(m1, load_bfrag(Wtf, (k * 2 + 1) * 4 + nt, lane), c[nt]);
    }
  }

  _Float16* L = lds[wid];
  // relu(t) -> LDS (wave-local; DS ops are in-order within a wave)
#pragma unroll
  for (int nt = 0; nt < 4; ++nt)
#pragma unroll
    for (int j = 0; j < 8; ++j) {
      float f = c[nt][j];
      f = f > 0.f ? f : 0.f;
      L[(j + 8 * hi) * 64 + nt * 16 + row] = (_Float16)f;
    }

  // ---- y = relu(t) @ W_d1 + b_d1 ----
  v16h ya0 = load_afrag(L + row * 64, hi);
  v16h ya1 = load_afrag(L + row * 64 + 32, hi);
  v8f c2[4];
#pragma unroll
  for (int nt = 0; nt < 4; ++nt) {
    float bv = bd1[nt * 16 + row];
#pragma unroll
    for (int j = 0; j < 8; ++j) c2[nt][j] = bv;
  }
#pragma unroll
  for (int nt = 0; nt < 4; ++nt) {
    c2[nt] = wmma_f16(ya0, load_bfrag(Wd1f, 0 * 4 + nt, lane), c2[nt]);
    c2[nt] = wmma_f16(ya1, load_bfrag(Wd1f, 1 * 4 + nt, lane), c2[nt]);
  }

  // relu(y) -> LDS
#pragma unroll
  for (int nt = 0; nt < 4; ++nt)
#pragma unroll
    for (int j = 0; j < 8; ++j) {
      float f = c2[nt][j];
      f = f > 0.f ? f : 0.f;
      L[(j + 8 * hi) * 64 + nt * 16 + row] = (_Float16)f;
    }

  // ---- z = relu(y) @ W_d2 + b_d2 ; out = z + xup ----
  v16h za0 = load_afrag(L + row * 64, hi);
  v16h za1 = load_afrag(L + row * 64 + 32, hi);
  v8f c3[2];
#pragma unroll
  for (int nt = 0; nt < 2; ++nt) {
    float bv = bd2[nt * 16 + row];
#pragma unroll
    for (int j = 0; j < 8; ++j) c3[nt][j] = bv;
  }
#pragma unroll
  for (int nt = 0; nt < 2; ++nt) {
    c3[nt] = wmma_f16(za0, load_bfrag(Wd2f, 0 * 2 + nt, lane), c3[nt]);
    c3[nt] = wmma_f16(za1, load_bfrag(Wd2f, 1 * 2 + nt, lane), c3[nt]);
  }
#pragma unroll
  for (int nt = 0; nt < 2; ++nt)
#pragma unroll
    for (int j = 0; j < 8; ++j) {
      int r = base + j + 8 * hi;
      if (r < M) {
        size_t o = (size_t)r * 32 + nt * 16 + row;
        out[o] = c3[nt][j] + xup[o];
      }
    }
}

// ---------------------------------------------------------------------------
extern "C" void kernel_launch(void* const* d_in, const int* in_sizes, int n_in,
                              void* d_out, int out_size, void* d_ws,
                              size_t ws_size, hipStream_t stream) {
  const float* feats = (const float*)d_in[0];
  const float* xup = (const float*)d_in[1];
  const int* nbr = (const int*)d_in[2];
  const int* parent = (const int*)d_in[3];
  const int* offs = (const int*)d_in[4];
  const float* W1 = (const float*)d_in[5];
  const float* b1 = (const float*)d_in[6];
  const float* W2 = (const float*)d_in[7];
  const float* b2 = (const float*)d_in[8];
  const float* W3 = (const float*)d_in[9];
  const float* b3 = (const float*)d_in[10];
  const float* Wt = (const float*)d_in[11];
  const float* bt = (const float*)d_in[12];
  const float* Wd1 = (const float*)d_in[13];
  const float* bd1 = (const float*)d_in[14];
  const float* Wd2 = (const float*)d_in[15];
  const float* bd2 = (const float*)d_in[16];
  float* out = (float*)d_out;

  const int N = in_sizes[0] / 32;   // coarse voxels
  const int M = in_sizes[1] / 32;   // fine voxels
  (void)n_in; (void)out_size; (void)ws_size;

  char* ws = (char*)d_ws;
  size_t woff = 0;
  auto take = [&](size_t bytes) -> void* {
    void* pp = ws + woff;
    woff = (woff + bytes + 255) & ~(size_t)255;
    return pp;
  };
  _Float16* Xf = (_Float16*)take((size_t)(N + 1) * 32 * 2);
  _Float16* HA = (_Float16*)take((size_t)(N + 1) * 64 * 2);
  _Float16* HB = (_Float16*)take((size_t)(N + 1) * 64 * 2);
  _Float16* HC = (_Float16*)take((size_t)(N + 1) * 64 * 2);
  _Float16* W1f = (_Float16*)take((size_t)27 * 1 * 4 * 512 * 2);
  _Float16* W2f = (_Float16*)take((size_t)27 * 2 * 4 * 512 * 2);
  _Float16* W3f = (_Float16*)take((size_t)27 * 2 * 4 * 512 * 2);
  _Float16* Wtf = (_Float16*)take((size_t)8 * 2 * 4 * 512 * 2);
  _Float16* Wd1f = (_Float16*)take((size_t)1 * 2 * 4 * 512 * 2);
  _Float16* Wd2f = (_Float16*)take((size_t)1 * 2 * 2 * 512 * 2);

  {
    size_t tot = (size_t)(N + 1) * 32;
    int blocks = (int)((tot + 255) / 256);
    if (blocks > 65535) blocks = 65535;
    cvt_feats_kernel<<<blocks, 256, 0, stream>>>(feats, Xf, N, 32);
  }
  zero_sentinel_kernel<<<1, 64, 0, stream>>>(HA + (size_t)N * 64,
                                             HB + (size_t)N * 64,
                                             HC + (size_t)N * 64);
  repack_weights_kernel<<<216, 256, 0, stream>>>(W1, W1f, 27, 32, 64);
  repack_weights_kernel<<<432, 256, 0, stream>>>(W2, W2f, 27, 64, 64);
  repack_weights_kernel<<<432, 256, 0, stream>>>(W3, W3f, 27, 64, 64);
  repack_weights_kernel<<<128, 256, 0, stream>>>(Wt, Wtf, 8, 64, 64);
  repack_weights_kernel<<<16, 256, 0, stream>>>(Wd1, Wd1f, 1, 64, 64);
  repack_weights_kernel<<<8, 256, 0, stream>>>(Wd2, Wd2f, 1, 64, 32);

  const int tilesN = (N + 15) / 16;
  const int blocksN =
      (tilesN + WAVES_PER_BLOCK * CONV_TM - 1) / (WAVES_PER_BLOCK * CONV_TM);
  conv3x3_kernel<32, CONV_TM><<<blocksN, 256, 0, stream>>>(Xf, nbr, W1f, b1, HA, N);
  conv3x3_kernel<64, CONV_TM><<<blocksN, 256, 0, stream>>>(HA, nbr, W2f, b2, HB, N);
  conv3x3_kernel<64, CONV_TM><<<blocksN, 256, 0, stream>>>(HB, nbr, W3f, b3, HC, N);

  const int tilesM = (M + 15) / 16;
  const int blocksM = (tilesM + WAVES_PER_BLOCK - 1) / WAVES_PER_BLOCK;
  upsample_decode_kernel<<<blocksM, 256, 0, stream>>>(
      HC, parent, offs, Wtf, bt, Wd1f, bd1, Wd2f, bd2, xup, out, M, N);
}
